// HeteGCNLayer_90005334655901
// MI455X (gfx1250) — matrix operand order repaired and checked by reading, not compiled, not run on hardware
//
#include <hip/hip_runtime.h>

typedef __attribute__((ext_vector_type(16))) _Float16 v16h;
typedef __attribute__((ext_vector_type(8)))  _Float16 v8h;
typedef __attribute__((ext_vector_type(8)))  float    v8f;
typedef __attribute__((ext_vector_type(4)))  float    v4f;

#define D 64

// ---------------------------------------------------------------------------
// Kernel 1: h = x @ w  (fp32 inner product over k=64).
// mode 0: store fp32 row-major (self-transform branch)
// mode 1: store f16 in WMMA B-fragment layout for v_wmma_f32_16x16x32_f16:
//   fragment block per (K-chunk kc of 32 rows, col-tile n of 16 cols) = 1024 B
//     bytes [   0..511]: v0..v3  -> lane*16 + vj*4 + half*2
//     bytes [512..1023]: v4..v7  -> 512 + lane*16 + (vj-4)*4 + half*2
//   lane = (k_local/16)*16 + col%16 ; vj = (k_local%16)/2 ; half = k_local%2
// ---------------------------------------------------------------------------
__global__ __launch_bounds__(256)
void hete_feat_kernel(const float* __restrict__ x,
                      const float* __restrict__ w,
                      void* __restrict__ out, int mode)
{
    __shared__ float xs[32][D];   // 8 KB
    __shared__ float ws[D][D];    // 16 KB
    const int t = threadIdx.x;
    const int rowbase = blockIdx.x * 32;

    for (int i = t; i < 32 * D; i += 256)
        xs[i >> 6][i & 63] = x[(size_t)rowbase * D + i];
    for (int i = t; i < D * D; i += 256)
        ws[i >> 6][i & 63] = w[i];
    __syncthreads();

    const int r  = t >> 3;         // 0..31 : local row (K index for B)
    const int c0 = (t & 7) << 3;   // 0,8,..,56 : first output column

    float acc[8] = {0.f, 0.f, 0.f, 0.f, 0.f, 0.f, 0.f, 0.f};
    for (int k = 0; k < D; ++k) {
        const float xv = xs[r][k];
#pragma unroll
        for (int j = 0; j < 8; ++j) acc[j] += xv * ws[k][c0 + j];
    }

    if (mode == 0) {
        float* o = (float*)out;
#pragma unroll
        for (int j = 0; j < 8; ++j)
            o[(size_t)(rowbase + r) * D + c0 + j] = acc[j];
    } else {
        _Float16* o = (_Float16*)out;
        const int kc   = blockIdx.x;       // K-chunk index (32 rows per chunk)
        const int kl   = r;                // k_local 0..31
        const int lgrp = (kl >> 4) << 4;   // lane group base (0 or 16)
        const int kk   = kl & 15;
        const int vj   = kk >> 1;
        const int half = kk & 1;
#pragma unroll
        for (int j = 0; j < 8; ++j) {
            const int c    = c0 + j;
            const int n    = c >> 4;          // col-tile
            const int lane = lgrp + (c & 15);
            const size_t halfoff = (size_t)(kc * 4 + n) * 512
                                 + (vj < 4 ? 0 : 256)
                                 + lane * 8 + (vj & 3) * 2 + half;
            o[halfoff] = (_Float16)acc[j];
        }
    }
}

// ---------------------------------------------------------------------------
// Kernel 2: out = (self + adj1 @ h1 + adj2 @ h2) / 3 + bias
// 128 threads = 4 wave32; wave w owns rows [blk*64 + w*16, +16), 64 cols.
// A (16x32 f16) built in-registers from fp32 adjacency (streamed once).
// B fragments loaded with 2 coalesced b128 loads per lane per col-tile.
// ---------------------------------------------------------------------------
__global__ __launch_bounds__(128)
void hete_agg_kernel(const float* __restrict__ adj1, int K1,
                     const _Float16* __restrict__ bf1,
                     const float* __restrict__ adj2, int K2,
                     const _Float16* __restrict__ bf2,
                     const float* __restrict__ selfft,
                     const float* __restrict__ bias,
                     float* __restrict__ out)
{
    const int lane = threadIdx.x & 31;
    const int wave = threadIdx.x >> 5;
    const int rowbase = blockIdx.x * 64 + wave * 16;
    const int m  = lane & 15;   // A-matrix row within tile
    const int hi = lane >> 4;   // lane group (0/1)

    v8f acc[4] = {};

#pragma unroll 1
    for (int rel = 0; rel < 2; ++rel) {
        const float*    adj = rel ? adj2 : adj1;
        const _Float16* bf  = rel ? bf2  : bf1;
        const int       K   = rel ? K2   : K1;
        // A f16 16x32 k order used here (same as B packing): lane<16 -> K {0..15}; lane>=16 -> K {16..31}
        const float* arow = adj + (size_t)(rowbase + m) * K + (hi << 4);
        const int nchunks = K >> 5;
        for (int kc = 0; kc < nchunks; ++kc) {
            const float* ap = arow + (kc << 5);
            __builtin_prefetch(ap + 32, 0, 0);   // global_prefetch_b8: next chunk
            const v4f f0 = *(const v4f*)(ap +  0);
            const v4f f1 = *(const v4f*)(ap +  4);
            const v4f f2 = *(const v4f*)(ap +  8);
            const v4f f3 = *(const v4f*)(ap + 12);
            v16h a;
#pragma unroll
            for (int i = 0; i < 4; ++i) {
                a[i]      = (_Float16)f0[i];   // K base+0..3   (v0,v1)
                a[4 + i]  = (_Float16)f1[i];   // K base+4..7   (v2,v3)
                a[8 + i]  = (_Float16)f2[i];   // K base+8..11  (v4,v5)
                a[12 + i] = (_Float16)f3[i];   // K base+12..15 (v6,v7)
            }

            const _Float16* fchunk = bf + (size_t)(kc << 2) * 512;
#pragma unroll
            for (int n = 0; n < 4; ++n) {
                const _Float16* tb = fchunk + n * 512;
                const v8h blo = *(const v8h*)(tb + lane * 8);        // v0..v3
                const v8h bhi = *(const v8h*)(tb + 256 + lane * 8);  // v4..v7
                v16h b;
#pragma unroll
                for (int i = 0; i < 8; ++i) { b[i] = blo[i]; b[8 + i] = bhi[i]; }
                acc[n] = __builtin_amdgcn_wmma_f32_16x16x32_f16(
                    false, a, false, b, (short)0, acc[n], false, false);
            }
        }
    }

    // Epilogue: mean over 3 branches + bias.
    // C/D layout: acc[i], lane l -> M = i + 8*(l/16), N = l%16
    const float scale = 1.0f / 3.0f;
#pragma unroll
    for (int n = 0; n < 4; ++n) {
        const int col = n * 16 + m;
        const float bv = bias[col];
#pragma unroll
        for (int i = 0; i < 8; ++i) {
            const size_t row = (size_t)(rowbase + i + (hi << 3));
            const size_t idx = row * D + col;
            out[idx] = (acc[n][i] + selfft[idx]) * scale + bv;
        }
    }
}

extern "C" void kernel_launch(void* const* d_in, const int* in_sizes, int n_in,
                              void* d_out, int out_size, void* d_ws, size_t ws_size,
                              hipStream_t stream) {
    const float* x_uav       = (const float*)d_in[0];
    const float* x_car       = (const float*)d_in[1];
    const float* x_poi       = (const float*)d_in[2];
    const float* adj_uav_car = (const float*)d_in[3];
    const float* adj_uav_poi = (const float*)d_in[4];
    const float* adj_car_uav = (const float*)d_in[5];
    const float* adj_car_poi = (const float*)d_in[6];
    const float* adj_poi_uav = (const float*)d_in[7];
    const float* adj_poi_car = (const float*)d_in[8];
    const float* w_self_uav  = (const float*)d_in[9];
    const float* w_self_car  = (const float*)d_in[10];
    const float* w_self_poi  = (const float*)d_in[11];
    const float* w_uav_car   = (const float*)d_in[12];
    const float* w_uav_poi   = (const float*)d_in[13];
    const float* w_car_uav   = (const float*)d_in[14];
    const float* w_car_poi   = (const float*)d_in[15];
    const float* w_poi_uav   = (const float*)d_in[16];
    const float* w_poi_car   = (const float*)d_in[17];
    const float* b_uav       = (const float*)d_in[18];
    const float* b_car       = (const float*)d_in[19];
    const float* b_poi       = (const float*)d_in[20];

    const int NU = 4096, NC = 4096, NP = 8192;

    char* ws = (char*)d_ws;
    float*    self_uav  = (float*)   (ws + 0);         // 1 MB
    float*    self_car  = (float*)   (ws + 1048576);   // 1 MB
    float*    self_poi  = (float*)   (ws + 2097152);   // 2 MB
    _Float16* f_uav_car = (_Float16*)(ws + 4194304);   // 512 KB
    _Float16* f_uav_poi = (_Float16*)(ws + 4718592);   // 1 MB
    _Float16* f_car_uav = (_Float16*)(ws + 5767168);   // 512 KB
    _Float16* f_car_poi = (_Float16*)(ws + 6291456);   // 1 MB
    _Float16* f_poi_uav = (_Float16*)(ws + 7340032);   // 512 KB
    _Float16* f_poi_car = (_Float16*)(ws + 7864320);   // 512 KB (total 8 MB)

    // Stage 1: per-type feature transforms (self fp32, relation f16 fragments)
    hete_feat_kernel<<<NU / 32, 256, 0, stream>>>(x_uav, w_self_uav, self_uav, 0);
    hete_feat_kernel<<<NC / 32, 256, 0, stream>>>(x_car, w_self_car, self_car, 0);
    hete_feat_kernel<<<NP / 32, 256, 0, stream>>>(x_poi, w_self_poi, self_poi, 0);
    hete_feat_kernel<<<NC / 32, 256, 0, stream>>>(x_car, w_uav_car, f_uav_car, 1);
    hete_feat_kernel<<<NP / 32, 256, 0, stream>>>(x_poi, w_uav_poi, f_uav_poi, 1);
    hete_feat_kernel<<<NU / 32, 256, 0, stream>>>(x_uav, w_car_uav, f_car_uav, 1);
    hete_feat_kernel<<<NP / 32, 256, 0, stream>>>(x_poi, w_car_poi, f_car_poi, 1);
    hete_feat_kernel<<<NU / 32, 256, 0, stream>>>(x_uav, w_poi_uav, f_poi_uav, 1);
    hete_feat_kernel<<<NC / 32, 256, 0, stream>>>(x_car, w_poi_car, f_poi_car, 1);

    // Stage 2: WMMA adjacency aggregation, fused mean + bias
    float* out = (float*)d_out;
    hete_agg_kernel<<<NU / 64, 128, 0, stream>>>(adj_uav_car, NC, f_uav_car,
                                                 adj_uav_poi, NP, f_uav_poi,
                                                 self_uav, b_uav, out);
    hete_agg_kernel<<<NC / 64, 128, 0, stream>>>(adj_car_uav, NU, f_car_uav,
                                                 adj_car_poi, NP, f_car_poi,
                                                 self_car, b_car, out + (size_t)NU * D);
    hete_agg_kernel<<<NP / 64, 128, 0, stream>>>(adj_poi_uav, NU, f_poi_uav,
                                                 adj_poi_car, NC, f_poi_car,
                                                 self_poi, b_poi, out + (size_t)(NU + NC) * D);
}
